// HANLayer_48344151884369
// MI455X (gfx1250) — matrix operand order repaired and checked by reading, not compiled, not run on hardware
//
#include <hip/hip_runtime.h>

// HAN layer for MI455X (gfx1250, wave32).
// GEMMs via split-bf16 (hi+lo) v_wmma_f32_16x16x32_bf16, fp32 accumulate.
// Edge softmax via native u32 atomic-max (monotone float encoding) + f32 atomics.

#define HN 50000      // nodes
#define HE 800000     // edges
#define FD 256        // input features
#define OD 256        // H*D
#define NHEAD 4
#define DHEAD 64

typedef __attribute__((ext_vector_type(16))) __bf16       v16bf;
typedef __attribute__((ext_vector_type(8)))  float        v8f;
typedef __attribute__((ext_vector_type(4)))  unsigned int v4u;

union FragBF { v4u u[2]; v16bf v; };

__device__ __forceinline__ unsigned short f32_to_bf16_rne(float x) {
  unsigned u = __float_as_uint(x);
  u += 0x7FFFu + ((u >> 16) & 1u);
  return (unsigned short)(u >> 16);
}
__device__ __forceinline__ float bf16hi_to_f32(unsigned short s) {
  return __uint_as_float(((unsigned)s) << 16);
}
// Monotone order-preserving float->uint encoding for atomicMax-based segment max.
__device__ __forceinline__ unsigned enc_f32(float x) {
  unsigned b = __float_as_uint(x);
  return (b & 0x80000000u) ? ~b : (b | 0x80000000u);
}
__device__ __forceinline__ float dec_f32(unsigned e) {
  unsigned b = (e & 0x80000000u) ? (e & 0x7FFFFFFFu) : ~e;
  return __uint_as_float(b);
}

// ---- f32 -> (bf16 hi, bf16 lo) split -------------------------------------
__global__ void k_split_bf16(const float* __restrict__ x,
                             unsigned short* __restrict__ hi,
                             unsigned short* __restrict__ lo, int n) {
  int i = blockIdx.x * blockDim.x + threadIdx.x;
  if (i >= n) return;
  float v = x[i];
  unsigned short h = f32_to_bf16_rne(v);
  hi[i] = h;
  lo[i] = f32_to_bf16_rne(v - bf16hi_to_f32(h));
}

// W [FD][OD] row-major -> transposed bf16 hi/lo [OD][FD]
__global__ void k_split_bf16_T(const float* __restrict__ W,
                               unsigned short* __restrict__ hiT,
                               unsigned short* __restrict__ loT) {
  int i = blockIdx.x * blockDim.x + threadIdx.x;  // i over FD*OD
  if (i >= FD * OD) return;
  int k = i >> 8;       // input-feature row (OD == 256)
  int c = i & 255;      // output column
  float v = W[i];
  unsigned short h = f32_to_bf16_rne(v);
  hiT[c * FD + k] = h;
  loT[c * FD + k] = f32_to_bf16_rne(v - bf16hi_to_f32(h));
}

// ---- Wh = h @ W : split-bf16 WMMA GEMM, C = [HN][OD] ---------------------
// Block = 256 threads = 8 waves covering a 32(M) x 64(N) region.
__global__ void k_gemm_wmma(const unsigned short* __restrict__ Ahi,
                            const unsigned short* __restrict__ Alo,
                            const unsigned short* __restrict__ Bhi,  // [OD][FD]
                            const unsigned short* __restrict__ Blo,
                            float* __restrict__ C) {
  const int wave = threadIdx.x >> 5;
  const int lane = threadIdx.x & 31;
  const int mt = blockIdx.x * 2 + (wave >> 2);
  const int nt = blockIdx.y * 4 + (wave & 3);
  if (mt * 16 >= HN) return;  // whole-wave uniform exit (EXEC stays all-1s for WMMA)

  const int row = mt * 16 + (lane & 15);
  const int col = nt * 16 + (lane & 15);
  const int kA = (lane >> 4) * 16;   // A 16x32: lanes 0-15 K=0..15, lanes 16-31 K=16..31
  const int kB = (lane >> 4) * 16;   // B 32x16: lanes 0-15 K=0..15, lanes 16-31 K=16..31

  const unsigned short* aH = Ahi + (size_t)row * FD;
  const unsigned short* aL = Alo + (size_t)row * FD;
  const unsigned short* bH = Bhi + (size_t)col * FD;
  const unsigned short* bL = Blo + (size_t)col * FD;

  v8f acc = {0.f, 0.f, 0.f, 0.f, 0.f, 0.f, 0.f, 0.f};
#pragma unroll
  for (int k0 = 0; k0 < FD; k0 += 32) {
    FragBF ah, al2, bh, bl;
    ah.u[0]  = *(const v4u*)(aH + k0 + kA);
    ah.u[1]  = *(const v4u*)(aH + k0 + kA + 8);
    al2.u[0] = *(const v4u*)(aL + k0 + kA);
    al2.u[1] = *(const v4u*)(aL + k0 + kA + 8);
    bh.u[0]  = *(const v4u*)(bH + k0 + kB);
    bh.u[1]  = *(const v4u*)(bH + k0 + kB + 8);
    bl.u[0]  = *(const v4u*)(bL + k0 + kB);
    bl.u[1]  = *(const v4u*)(bL + k0 + kB + 8);
    // fp32 emulation: hi*hi + hi*lo + lo*hi (lo*lo below fp32 ulp)
    acc = __builtin_amdgcn_wmma_f32_16x16x32_bf16(false, ah.v,  false, bh.v, (short)0, acc, false, false);
    acc = __builtin_amdgcn_wmma_f32_16x16x32_bf16(false, ah.v,  false, bl.v, (short)0, acc, false, false);
    acc = __builtin_amdgcn_wmma_f32_16x16x32_bf16(false, al2.v, false, bh.v, (short)0, acc, false, false);
  }
  // C layout: VGPR v -> M = (lane>>4)*8 + v, N = lane&15
  float* crow = C + (size_t)(mt * 16 + (lane >> 4) * 8) * OD + col;
#pragma unroll
  for (int v = 0; v < 8; ++v) crow[(size_t)v * OD] = acc[v];
}

// ---- el/er: per-node per-head dot(Wh[n,h,:], al/ar[h,:]) -----------------
__global__ void k_attn_coef(const float* __restrict__ Wh, const float* __restrict__ al,
                            const float* __restrict__ ar, float* __restrict__ el,
                            float* __restrict__ er) {
  __shared__ float sl[256], sr[256];
  int n = blockIdx.x, t = threadIdx.x;
  float w = Wh[(size_t)n * OD + t];
  sl[t] = w * al[t];
  sr[t] = w * ar[t];
  for (int sft = 32; sft >= 1; sft >>= 1) {
    __syncthreads();
    if ((t & 63) < sft) { sl[t] += sl[t + sft]; sr[t] += sr[t + sft]; }
  }
  __syncthreads();
  if ((t & 63) == 0) {
    el[n * NHEAD + (t >> 6)] = sl[t];
    er[n * NHEAD + (t >> 6)] = sr[t];
  }
}

__global__ void k_zero(float* __restrict__ p, int n) {
  int i = blockIdx.x * blockDim.x + threadIdx.x;
  if (i < n) p[i] = 0.f;
}

// ---- edge pass 1: segment max of leaky_relu(el[src]+er[dst]) -------------
__global__ void k_edge_max(const float* __restrict__ el, const float* __restrict__ er,
                           const int* __restrict__ src, const int* __restrict__ dst,
                           unsigned* __restrict__ menc) {
  int g = blockIdx.x * blockDim.x + threadIdx.x;
  if (g >= HE * NHEAD) return;
  int e = g >> 2, h = g & 3;
  int s = src[e], d = dst[e];
  float v = el[s * NHEAD + h] + er[d * NHEAD + h];
  v = v > 0.f ? v : 0.2f * v;
  atomicMax(&menc[d * NHEAD + h], enc_f32(v));
}

// Decode in place: enc==0 sentinel -> 0.0 (matches where(isfinite(m), m, 0))
__global__ void k_dec_max(unsigned* __restrict__ menc, int n) {
  int i = blockIdx.x * blockDim.x + threadIdx.x;
  if (i >= n) return;
  unsigned e = menc[i];
  ((float*)menc)[i] = (e == 0u) ? 0.f : dec_f32(e);
}

// ---- edge pass 2: ex = exp(e - m[dst]); s[dst] += ex ---------------------
__global__ void k_edge_exp(const float* __restrict__ el, const float* __restrict__ er,
                           const int* __restrict__ src, const int* __restrict__ dst,
                           const float* __restrict__ m, float* __restrict__ ssum,
                           float* __restrict__ ex) {
  int g = blockIdx.x * blockDim.x + threadIdx.x;
  if (g >= HE * NHEAD) return;
  int e = g >> 2, h = g & 3;
  int s = src[e], d = dst[e];
  float v = el[s * NHEAD + h] + er[d * NHEAD + h];
  v = v > 0.f ? v : 0.2f * v;
  float x = expf(v - m[d * NHEAD + h]);
  ex[g] = x;  // [E][H] flat == g
  atomicAdd(&ssum[d * NHEAD + h], x);
}

// ---- edge pass 3: z[dst] += (ex/s[dst]) * Wh[src], one block per edge ----
__global__ void k_scatter(const float* __restrict__ Wh, const float* __restrict__ ex,
                          const float* __restrict__ ssum, const int* __restrict__ src,
                          const int* __restrict__ dst, float* __restrict__ z) {
  int e = blockIdx.x;
  int t = threadIdx.x;    // 0..255, h = t>>6, d = t&63
  int h = t >> 6;
  int s = src[e], d = dst[e];
  float alpha = ex[e * NHEAD + h] / ssum[d * NHEAD + h];
  float val = alpha * Wh[(size_t)s * OD + t];
  atomicAdd(&z[(size_t)d * OD + t], val);
}

// ---- z = elu(z + b) ------------------------------------------------------
__global__ void k_finalize(float* __restrict__ z, const float* __restrict__ b) {
  int i = blockIdx.x * blockDim.x + threadIdx.x;
  if (i >= HN * OD) return;
  float v = z[i] + b[i & 255];
  z[i] = v > 0.f ? v : (expf(v) - 1.f);
}

// ---- semantic attention over P=2 meta-paths, one block (128 thr)/node ----
__global__ void k_semantic(const float* __restrict__ z0, const float* __restrict__ z1,
                           const float* __restrict__ Wp1, const float* __restrict__ bp1,
                           const float* __restrict__ Wp2, float* __restrict__ out) {
  __shared__ float zz[2][OD];
  __shared__ float red[128];
  __shared__ float wsc[2];
  int n = blockIdx.x, t = threadIdx.x;  // 128 threads
  zz[0][t]       = z0[(size_t)n * OD + t];
  zz[0][t + 128] = z0[(size_t)n * OD + t + 128];
  zz[1][t]       = z1[(size_t)n * OD + t];
  zz[1][t + 128] = z1[(size_t)n * OD + t + 128];
  __syncthreads();
  for (int p = 0; p < 2; ++p) {
    float acc = bp1[t];
#pragma unroll 8
    for (int k = 0; k < OD; ++k) acc += zz[p][k] * Wp1[k * 128 + t];
    red[t] = tanhf(acc) * Wp2[t];
    for (int sft = 64; sft >= 1; sft >>= 1) {
      __syncthreads();
      if (t < sft) red[t] += red[t + sft];
    }
    __syncthreads();
    if (t == 0) wsc[p] = red[0];
    __syncthreads();
  }
  float w0 = wsc[0], w1 = wsc[1];
  float mw = fmaxf(w0, w1);
  float e0 = expf(w0 - mw), e1 = expf(w1 - mw);
  float inv = 1.f / (e0 + e1);
  float b0 = e0 * inv, b1 = e1 * inv;
  out[(size_t)n * OD + t]       = b0 * zz[0][t] + b1 * zz[1][t];
  out[(size_t)n * OD + t + 128] = b0 * zz[0][t + 128] + b1 * zz[1][t + 128];
}

extern "C" void kernel_launch(void* const* d_in, const int* in_sizes, int n_in,
                              void* d_out, int out_size, void* d_ws, size_t ws_size,
                              hipStream_t stream) {
  (void)in_sizes; (void)n_in; (void)out_size; (void)ws_size;
  const float* h   = (const float*)d_in[0];
  const int* src0  = (const int*)d_in[1];
  const int* dst0  = (const int*)d_in[2];
  const int* src1  = (const int*)d_in[3];
  const int* dst1  = (const int*)d_in[4];
  const float* W0  = (const float*)d_in[5];
  const float* al0 = (const float*)d_in[6];
  const float* ar0 = (const float*)d_in[7];
  const float* b0  = (const float*)d_in[8];
  const float* W1  = (const float*)d_in[9];
  const float* al1 = (const float*)d_in[10];
  const float* ar1 = (const float*)d_in[11];
  const float* b1  = (const float*)d_in[12];
  const float* Wp1 = (const float*)d_in[13];
  const float* bp1 = (const float*)d_in[14];
  const float* Wp2 = (const float*)d_in[15];
  float* out = (float*)d_out;

  // Workspace carve-up (256B aligned). Layers processed sequentially so
  // Wh / el / er / menc / ssum / ex / Bt buffers are reused. Total ~221 MB.
  char* ws = (char*)d_ws;
  size_t off = 0;
  auto carve = [&](size_t bytes) -> void* {
    void* p = ws + off;
    off += (bytes + 255) & ~(size_t)255;
    return p;
  };
  unsigned short* Ahi = (unsigned short*)carve((size_t)HN * FD * 2);
  unsigned short* Alo = (unsigned short*)carve((size_t)HN * FD * 2);
  unsigned short* Bhi = (unsigned short*)carve((size_t)FD * OD * 2);
  unsigned short* Blo = (unsigned short*)carve((size_t)FD * OD * 2);
  float* Wh   = (float*)carve((size_t)HN * OD * 4);
  float* z0   = (float*)carve((size_t)HN * OD * 4);
  float* z1   = (float*)carve((size_t)HN * OD * 4);
  float* el   = (float*)carve((size_t)HN * NHEAD * 4);
  float* er   = (float*)carve((size_t)HN * NHEAD * 4);
  unsigned* menc = (unsigned*)carve((size_t)HN * NHEAD * 4);
  float* ssum = (float*)carve((size_t)HN * NHEAD * 4);
  float* ex   = (float*)carve((size_t)HE * NHEAD * 4);

  const int T = 256;
  k_split_bf16<<<(HN * FD + T - 1) / T, T, 0, stream>>>(h, Ahi, Alo, HN * FD);
  k_zero<<<(HN * OD + T - 1) / T, T, 0, stream>>>(z0, HN * OD);
  k_zero<<<(HN * OD + T - 1) / T, T, 0, stream>>>(z1, HN * OD);

  const int MT2 = ((HN + 15) / 16 + 1) / 2;  // M-tile pairs: 1563
  for (int layer = 0; layer < 2; ++layer) {
    const float* W  = layer ? W1 : W0;
    const float* al = layer ? al1 : al0;
    const float* ar = layer ? ar1 : ar0;
    const float* bb = layer ? b1 : b0;
    const int* src  = layer ? src1 : src0;
    const int* dst  = layer ? dst1 : dst0;
    float* z        = layer ? z1 : z0;

    k_split_bf16_T<<<(FD * OD + T - 1) / T, T, 0, stream>>>(W, Bhi, Blo);
    k_gemm_wmma<<<dim3(MT2, 4), 256, 0, stream>>>(Ahi, Alo, Bhi, Blo, Wh);
    k_attn_coef<<<HN, 256, 0, stream>>>(Wh, al, ar, el, er);
    k_zero<<<(HN * NHEAD + T - 1) / T, T, 0, stream>>>((float*)menc, HN * NHEAD);
    k_zero<<<(HN * NHEAD + T - 1) / T, T, 0, stream>>>(ssum, HN * NHEAD);
    k_edge_max<<<(HE * NHEAD + T - 1) / T, T, 0, stream>>>(el, er, src, dst, menc);
    k_dec_max<<<(HN * NHEAD + T - 1) / T, T, 0, stream>>>(menc, HN * NHEAD);
    k_edge_exp<<<(HE * NHEAD + T - 1) / T, T, 0, stream>>>(el, er, src, dst,
                                                           (const float*)menc, ssum, ex);
    k_scatter<<<HE, 256, 0, stream>>>(Wh, ex, ssum, src, dst, z);
    k_finalize<<<(HN * OD + T - 1) / T, T, 0, stream>>>(z, bb);
  }
  k_semantic<<<HN, 128, 0, stream>>>(z0, z1, Wp1, bp1, Wp2, out);
}
